// Decoder_42425686950076
// MI455X (gfx1250) — matrix operand order repaired and checked by reading, not compiled, not run on hardware
//
#include <hip/hip_runtime.h>

// ---------------------------------------------------------------------------
// Transformer decoder for MI455X (gfx1250, wave32, WMMA).
// All GEMMs + attention use v_wmma_f32_16x16x32_bf16 (f32 accumulate).
// GEMM is double-buffered: GLOBAL_LOAD_ASYNC_TO_LDS_B128 for tile i+1 is in
// flight while tile i computes; s_wait_asynccnt 2 separates the pipeline
// stages (async loads complete in order per the CDNA5 ISA).
// ---------------------------------------------------------------------------

typedef unsigned short u16;
typedef unsigned int   u32;
typedef __attribute__((ext_vector_type(16))) __bf16 v16bf;
typedef __attribute__((ext_vector_type(8)))  float  v8f;
typedef __attribute__((ext_vector_type(4)))  int    v4i;

#define DEV __device__ __forceinline__

#if __has_builtin(__builtin_amdgcn_global_load_async_to_lds_b128)
#define HAVE_ASYNC_LDS 1
#else
#define HAVE_ASYNC_LDS 0
#endif

#if HAVE_ASYNC_LDS
// builtin takes pointers to 4 x i32 vectors; global/LDS address spaces are
// inferred from pointer provenance (confirmed by round-2 diagnostics).
#define ASYNC_PTR(p) ((v4i*)(void*)(p))
DEV void async_wait0() {
#if __has_builtin(__builtin_amdgcn_s_wait_asynccnt)
  __builtin_amdgcn_s_wait_asynccnt(0);
#else
  asm volatile("s_wait_asynccnt 0x0" ::: "memory");
#endif
}
DEV void async_wait2() {   // allow 2 newest async loads to remain in flight
#if __has_builtin(__builtin_amdgcn_s_wait_asynccnt)
  __builtin_amdgcn_s_wait_asynccnt(2);
#else
  asm volatile("s_wait_asynccnt 0x2" ::: "memory");
#endif
}
#endif

constexpr int Bb  = 32;
constexpr int Ss  = 256;
constexpr int SEe = 1024;
constexpr int Dd  = 512;
constexpr int Vv  = 8000;
constexpr int Ll  = 4;
constexpr int Hh  = 8;      // heads, dk = 64

union Frag {               // one WMMA 16x16x32 bf16 operand (8 VGPRs)
  uint4 q[2];
  u32   u[8];
  v16bf v;
};

DEV u16 f2bf(float f) {     // f32 -> bf16, round-to-nearest-even
  u32 u = __float_as_uint(f);
  u32 r = u + 0x7FFFu + ((u >> 16) & 1u);
  return (u16)(r >> 16);
}

DEV v8f wmma_bf16(const Frag& a, const Frag& b, v8f c) {
  return __builtin_amdgcn_wmma_f32_16x16x32_bf16(false, a.v, false, b.v,
                                                 (short)0, c, false, false);
}

// ---------------------------------------------------------------------------
// f32 -> bf16 elementwise conversion
// ---------------------------------------------------------------------------
__global__ void cvt_f32_bf16(const float* __restrict__ in, u16* __restrict__ out, int n) {
  int i = blockIdx.x * blockDim.x + threadIdx.x;
  int stride = gridDim.x * blockDim.x;
  for (; i < n; i += stride) out[i] = f2bf(in[i]);
}

// ---------------------------------------------------------------------------
// x = emb[seq] + pos  ->  xF (f32) and xB (bf16)
// ---------------------------------------------------------------------------
__global__ void embed_kernel(const int* __restrict__ seq, const float* __restrict__ emb,
                             const float* __restrict__ pos, float* __restrict__ xF,
                             u16* __restrict__ xB) {
  int r = blockIdx.x;                 // 0 .. B*S-1
  int s = r & (Ss - 1);
  int tok = seq[r];
  for (int c = threadIdx.x; c < Dd; c += blockDim.x) {
    float v = emb[(size_t)tok * Dd + c] + pos[(size_t)s * Dd + c];
    xF[(size_t)r * Dd + c] = v;
    xB[(size_t)r * Dd + c] = f2bf(v);
  }
}

// ---------------------------------------------------------------------------
// Tiled WMMA GEMM:  Y[M,N] = X_b16[M,K] @ W_b16[K,N] + bias  (+ epilogue)
//   EPI 0: bias -> bf16 out                      (Q/K/V projections)
//   EPI 1: bias + residual(f32) + BN -> f32+bf16 (Wo / FFN2 + BN)
//   EPI 2: bias + relu -> bf16                   (FFN1)
//   EPI 3: bias -> f32, N edge-guarded           (logits, N=8000)
// Block: 256 threads = 8 waves; tile 128x128; wave = 32(M) x 64(N); K step 32.
// Double-buffered LDS (buf i computes while buf i^1 fills asynchronously).
// ---------------------------------------------------------------------------
template <int EPI>
__global__ __launch_bounds__(256)
void gemm_kernel(const u16* __restrict__ X, const u16* __restrict__ W,
                 const float* __restrict__ bias, const float* __restrict__ res,
                 const float* __restrict__ gamma, const float* __restrict__ beta,
                 const float* __restrict__ mean, const float* __restrict__ var,
                 float* __restrict__ outF, u16* __restrict__ outB,
                 int M, int N, int K) {
  __shared__ __align__(16) u16 lA[2][128 * 32];   // [m][k]
  __shared__ __align__(16) u16 lB[2][128 * 32];   // transposed: [n][k]

  const int tid  = threadIdx.x;
  const int wid  = tid >> 5, lane = tid & 31;
  const int wm   = wid >> 1, wn = wid & 1;
  const int ln   = lane & 15, hf = lane >> 4;
  const int mb   = blockIdx.x * 128;
  const int nb   = blockIdx.y * 128;

  // staging indices: A tile rows (ar0, ar0+64), B tile K-pair kp / col-seg ns
  const int ar0 = tid >> 2, as0 = tid & 3;
  const int kp  = tid >> 4, ns  = tid & 15;

  auto stageA = [&](int bufi, int ktv) {
#if HAVE_ASYNC_LDS
    __builtin_amdgcn_global_load_async_to_lds_b128(
        ASYNC_PTR(X + (size_t)(mb + ar0) * K + ktv + as0 * 8),
        ASYNC_PTR(&lA[bufi][ar0 * 32 + as0 * 8]), 0, 0);
    __builtin_amdgcn_global_load_async_to_lds_b128(
        ASYNC_PTR(X + (size_t)(mb + ar0 + 64) * K + ktv + as0 * 8),
        ASYNC_PTR(&lA[bufi][(ar0 + 64) * 32 + as0 * 8]), 0, 0);
#else
    *(uint4*)&lA[bufi][ar0 * 32 + as0 * 8] =
        *(const uint4*)(X + (size_t)(mb + ar0) * K + ktv + as0 * 8);
    *(uint4*)&lA[bufi][(ar0 + 64) * 32 + as0 * 8] =
        *(const uint4*)(X + (size_t)(mb + ar0 + 64) * K + ktv + as0 * 8);
#endif
  };
  auto loadB = [&](int ktv, uint4& t0, uint4& t1) {
    int col0 = nb + ns * 8;          // N and col0 are multiples of 8
    if (col0 + 8 <= N) {
      t0 = *(const uint4*)(W + (size_t)(ktv + 2 * kp) * N + col0);
      t1 = *(const uint4*)(W + (size_t)(ktv + 2 * kp + 1) * N + col0);
    } else {
      t0 = make_uint4(0u, 0u, 0u, 0u);
      t1 = make_uint4(0u, 0u, 0u, 0u);
    }
  };
  auto storeB = [&](int bufi, uint4 t0, uint4 t1) {
    union { uint4 q; u16 s[8]; } a, b;
    a.q = t0; b.q = t1;
#pragma unroll
    for (int e = 0; e < 8; e++)
      *(u32*)&lB[bufi][(ns * 8 + e) * 32 + 2 * kp] =
          (u32)a.s[e] | ((u32)b.s[e] << 16);
  };

  const v8f zf = {0.f, 0.f, 0.f, 0.f, 0.f, 0.f, 0.f, 0.f};
  v8f acc[2][4];
#pragma unroll
  for (int r = 0; r < 2; r++)
#pragma unroll
    for (int c = 0; c < 4; c++) acc[r][c] = zf;

  // ---- prologue: stage tile 0 into buffer 0 ----
  uint4 nb0, nb1;
  stageA(0, 0);
  loadB(0, nb0, nb1);
  storeB(0, nb0, nb1);

  int buf = 0;
  for (int kt = 0; kt < K; kt += 32, buf ^= 1) {
    const bool hn = (kt + 32) < K;     // uniform across block
    if (hn) {
      stageA(buf ^ 1, kt + 32);        // async fill of the other buffer
      loadB(kt + 32, nb0, nb1);        // global loads overlap compute below
      __builtin_prefetch(X + (size_t)(mb + ar0) * K + kt + 64, 0, 1);
    }
#if HAVE_ASYNC_LDS
    if (hn) async_wait2(); else async_wait0();   // tile kt's A has landed
#endif
    __syncthreads();                   // A[buf] + B[buf] visible to all waves

    Frag af[2], bf[4];
#pragma unroll
    for (int r = 0; r < 2; r++) {
      int base = (wm * 32 + r * 16 + ln) * 32 + hf * 8;
      af[r].q[0] = *(const uint4*)&lA[buf][base];
      af[r].q[1] = *(const uint4*)&lA[buf][base + 16];
    }
#pragma unroll
    for (int c = 0; c < 4; c++) {
      int base = (wn * 64 + c * 16 + ln) * 32 + hf * 8;
      bf[c].q[0] = *(const uint4*)&lB[buf][base];
      bf[c].q[1] = *(const uint4*)&lB[buf][base + 16];
    }
#pragma unroll
    for (int r = 0; r < 2; r++)
#pragma unroll
      for (int c = 0; c < 4; c++) acc[r][c] = wmma_bf16(af[r], bf[c], acc[r][c]);

    if (hn) storeB(buf ^ 1, nb0, nb1); // fill other buffer (no readers yet)
    __syncthreads();                   // all reads of buf done before reuse
  }

  // ------------------------------ epilogue ---------------------------------
#pragma unroll
  for (int r = 0; r < 2; r++) {
#pragma unroll
    for (int c = 0; c < 4; c++) {
      int col = nb + wn * 64 + c * 16 + ln;
      float bia = (col < N) ? bias[col] : 0.f;
      float g = 0.f, bt = 0.f, mn = 0.f, iv = 0.f;
      if constexpr (EPI == 1) {
        g = gamma[col]; bt = beta[col]; mn = mean[col];
        iv = rsqrtf(var[col] + 1e-3f);
      }
#pragma unroll
      for (int j = 0; j < 8; j++) {
        int row = mb + wm * 32 + r * 16 + (hf ? j + 8 : j);
        float v = acc[r][c][j] + bia;
        if constexpr (EPI == 0) {
          outB[(size_t)row * N + col] = f2bf(v);
        } else if constexpr (EPI == 1) {
          float t = res[(size_t)row * N + col] + v;
          float o = g * (t - mn) * iv + bt;
          outF[(size_t)row * N + col] = o;
          outB[(size_t)row * N + col] = f2bf(o);
        } else if constexpr (EPI == 2) {
          v = fmaxf(v, 0.f);
          outB[(size_t)row * N + col] = f2bf(v);
        } else {
          if (col < N) outF[(size_t)row * N + col] = v;
        }
      }
    }
  }
}

// ---------------------------------------------------------------------------
// Flash attention (WMMA, online softmax). dk = 64, heads concatenated in D.
// Grid: (B*H, Sq/64), block 128 = 4 waves; each wave owns 16 query rows.
// Key tiles of 32. CAUSAL uses the reference's STRICT mask (key < query),
// so query row 0 yields exactly zero (run_s == 0 guard).
// ---------------------------------------------------------------------------
template <bool CAUSAL>
__global__ __launch_bounds__(128)
void attn_kernel(const u16* __restrict__ Q, const u16* __restrict__ Kp,
                 const u16* __restrict__ Vp, u16* __restrict__ O,
                 int Sq, int Sk) {
  __shared__ __align__(16) u16 lK[32 * 64];      // [key][d]
  __shared__ __align__(16) u16 lV[64 * 32];      // transposed: [d][key]
  __shared__ __align__(16) u16 lP[4][16 * 32];   // per-wave P scratch [m][k]

  const int b  = blockIdx.x >> 3;                // / H
  const int h  = blockIdx.x & 7;                 // % H
  const int qb = blockIdx.y * 64;
  const int tid = threadIdx.x;
  const int w = tid >> 5, lane = tid & 31;
  const int ln = lane & 15, hf = lane >> 4;

  const v8f zf = {0.f, 0.f, 0.f, 0.f, 0.f, 0.f, 0.f, 0.f};

  // Q fragments (rows qb + w*16 + m), two 16x32 chunks over dk=64
  Frag aq[2];
  {
    size_t row = (size_t)b * Sq + qb + w * 16 + ln;
    const u16* qp = Q + row * Dd + h * 64;
#pragma unroll
    for (int ch = 0; ch < 2; ch++) {
      aq[ch].q[0] = *(const uint4*)(qp + ch * 32 + hf * 8);
      aq[ch].q[1] = *(const uint4*)(qp + ch * 32 + 16 + hf * 8);
    }
  }

  v8f co[4];
#pragma unroll
  for (int c = 0; c < 4; c++) co[c] = zf;
  float run_m[8], run_s[8];
#pragma unroll
  for (int j = 0; j < 8; j++) { run_m[j] = -1e30f; run_s[j] = 0.f; }

  const float scale = 0.125f;   // 1/sqrt(64)

  int kend = Sk;
  if (CAUSAL) { int lim = qb + 64; kend = lim < Sk ? lim : Sk; }

  for (int kt = 0; kt < kend; kt += 32) {
    // stage K tile [32 keys][64]
#if HAVE_ASYNC_LDS
#pragma unroll
    for (int it = 0; it < 2; it++) {
      int i = tid + it * 128;
      int key = i >> 3, seg = i & 7;
      __builtin_amdgcn_global_load_async_to_lds_b128(
          ASYNC_PTR(Kp + ((size_t)b * Sk + kt + key) * Dd + h * 64 + seg * 8),
          ASYNC_PTR(&lK[key * 64 + seg * 8]), 0, 0);
    }
#else
#pragma unroll
    for (int it = 0; it < 2; it++) {
      int i = tid + it * 128;
      int key = i >> 3, seg = i & 7;
      *(uint4*)&lK[key * 64 + seg * 8] =
          *(const uint4*)(Kp + ((size_t)b * Sk + kt + key) * Dd + h * 64 + seg * 8);
    }
#endif
    // stage V tile transposed [d][key]: two keys per item, packed b32 stores
#pragma unroll
    for (int it = 0; it < 4; it++) {
      int i = tid + it * 128;
      int key = (i & 15) * 2, dp = i >> 4;     // keys (key,key+1), d = 2*dp
      const u16* vp = Vp + ((size_t)b * Sk + kt + key) * Dd + h * 64 + 2 * dp;
      u32 u0 = *(const u32*)vp;
      u32 u1 = *(const u32*)(vp + Dd);
      *(u32*)&lV[(2 * dp) * 32 + key]     = (u0 & 0xffffu) | (u1 << 16);
      *(u32*)&lV[(2 * dp + 1) * 32 + key] = (u0 >> 16) | (u1 & 0xffff0000u);
    }
#if HAVE_ASYNC_LDS
    async_wait0();
#endif
    __syncthreads();

    // scores: two 16x16 tiles (keys kt..kt+15, kt+16..kt+31)
    v8f svec[2];
#pragma unroll
    for (int t = 0; t < 2; t++) {
      Frag bk0, bk1;
      int base0 = (t * 16 + ln) * 64 + hf * 8;
      bk0.q[0] = *(const uint4*)&lK[base0];
      bk0.q[1] = *(const uint4*)&lK[base0 + 16];
      int base1 = (t * 16 + ln) * 64 + 32 + hf * 8;
      bk1.q[0] = *(const uint4*)&lK[base1];
      bk1.q[1] = *(const uint4*)&lK[base1 + 16];
      v8f sc = wmma_bf16(aq[0], bk0, zf);
      sc = wmma_bf16(aq[1], bk1, sc);
      svec[t] = sc;
    }

    // online softmax (row reductions across the 16-lane halves)
#pragma unroll
    for (int j = 0; j < 8; j++) {
      int mrow = hf ? j + 8 : j;
      int qg = qb + w * 16 + mrow;
      float x0 = svec[0][j] * scale, x1 = svec[1][j] * scale;
      bool v0 = !CAUSAL || (kt + ln) < qg;
      bool v1 = !CAUSAL || (kt + 16 + ln) < qg;
      if (!v0) x0 = -1e30f;
      if (!v1) x1 = -1e30f;
      float rm = fmaxf(x0, x1);
#pragma unroll
      for (int msk = 1; msk < 16; msk <<= 1) rm = fmaxf(rm, __shfl_xor(rm, msk, 16));
      float nm = fmaxf(run_m[j], rm);
      float p0 = v0 ? __expf(x0 - nm) : 0.f;
      float p1 = v1 ? __expf(x1 - nm) : 0.f;
      float rs = p0 + p1;
#pragma unroll
      for (int msk = 1; msk < 16; msk <<= 1) rs += __shfl_xor(rs, msk, 16);
      float corr = __expf(run_m[j] - nm);
      run_m[j] = nm;
      run_s[j] = run_s[j] * corr + rs;
#pragma unroll
      for (int c = 0; c < 4; c++) co[c][j] *= corr;
      lP[w][mrow * 32 + ln]      = f2bf(p0);
      lP[w][mrow * 32 + 16 + ln] = f2bf(p1);
    }

    // P(16x32) @ V(32x64) accumulated into co (same-wave LDS round trip)
    Frag pa;
    {
      int base = ln * 32 + hf * 8;
      pa.q[0] = *(const uint4*)&lP[w][base];
      pa.q[1] = *(const uint4*)&lP[w][base + 16];
    }
#pragma unroll
    for (int c = 0; c < 4; c++) {
      Frag bv;
      int base = (c * 16 + ln) * 32 + hf * 8;
      bv.q[0] = *(const uint4*)&lV[base];
      bv.q[1] = *(const uint4*)&lV[base + 16];
      co[c] = wmma_bf16(pa, bv, co[c]);
    }
    __syncthreads();
  }

  // normalize + write out (bf16), zero rows with no valid keys
#pragma unroll
  for (int c = 0; c < 4; c++)
#pragma unroll
    for (int j = 0; j < 8; j++) {
      int mrow = hf ? j + 8 : j;
      size_t row = (size_t)b * Sq + qb + w * 16 + mrow;
      float denom = run_s[j];
      float val = denom > 0.f ? co[c][j] / denom : 0.f;
      O[row * Dd + h * 64 + c * 16 + ln] = f2bf(val);
    }
}

// ---------------------------------------------------------------------------
// Host orchestration
// ---------------------------------------------------------------------------
extern "C" void kernel_launch(void* const* d_in, const int* in_sizes, int n_in,
                              void* d_out, int out_size, void* d_ws, size_t ws_size,
                              hipStream_t stream) {
  (void)in_sizes; (void)n_in; (void)out_size; (void)ws_size;

  const int*   seq  = (const int*)  d_in[0];
  const float* enc  = (const float*)d_in[1];
  const float* pos  = (const float*)d_in[2];
  const float* emb  = (const float*)d_in[3];
  const float* Wout = (const float*)d_in[4];
  const float* bout = (const float*)d_in[5];
  const float* W1   = (const float*)d_in[6];
  const float* b1   = (const float*)d_in[7];
  const float* W2   = (const float*)d_in[8];
  const float* b2   = (const float*)d_in[9];
  const float* Wq_bot = (const float*)d_in[10];
  const float* Wk_bot = (const float*)d_in[11];
  const float* Wv_bot = (const float*)d_in[12];
  const float* Wo_bot = (const float*)d_in[13];
  const float* bq_bot = (const float*)d_in[14];
  const float* bk_bot = (const float*)d_in[15];
  const float* bv_bot = (const float*)d_in[16];
  const float* bo_bot = (const float*)d_in[17];
  const float* Wq_mid = (const float*)d_in[18];
  const float* Wk_mid = (const float*)d_in[19];
  const float* Wv_mid = (const float*)d_in[20];
  const float* Wo_mid = (const float*)d_in[21];
  const float* bq_mid = (const float*)d_in[22];
  const float* bk_mid = (const float*)d_in[23];
  const float* bv_mid = (const float*)d_in[24];
  const float* bo_mid = (const float*)d_in[25];
  const float* gamma_bot = (const float*)d_in[26];
  const float* beta_bot  = (const float*)d_in[27];
  const float* mean_bot  = (const float*)d_in[28];
  const float* var_bot   = (const float*)d_in[29];
  const float* gamma_mid = (const float*)d_in[30];
  const float* beta_mid  = (const float*)d_in[31];
  const float* mean_mid  = (const float*)d_in[32];
  const float* var_mid   = (const float*)d_in[33];
  const float* gamma_ffn = (const float*)d_in[34];
  const float* beta_ffn  = (const float*)d_in[35];
  const float* mean_ffn  = (const float*)d_in[36];
  const float* var_ffn   = (const float*)d_in[37];

  // ---- workspace carve (256B aligned), ~180 MB total ----
  char* basep = (char*)d_ws;
  size_t off = 0;
  auto carve = [&](size_t bytes) -> char* {
    char* r = basep + off;
    off += (bytes + 255) & ~(size_t)255;
    return r;
  };
  u16*   encB = (u16*)  carve((size_t)Bb * SEe * Dd * 2);
  float* xF   = (float*)carve((size_t)Bb * Ss * Dd * 4);
  u16*   xB   = (u16*)  carve((size_t)Bb * Ss * Dd * 2);
  u16*   qB   = (u16*)  carve((size_t)Bb * Ss * Dd * 2);
  u16*   kB   = (u16*)  carve((size_t)Bb * SEe * Dd * 2);   // big enough for cross
  u16*   vB   = (u16*)  carve((size_t)Bb * SEe * Dd * 2);
  u16*   aB   = (u16*)  carve((size_t)Bb * Ss * Dd * 2);
  u16*   mB   = (u16*)  carve((size_t)Bb * Ss * 512 * 2);
  u16*   wB   = (u16*)  carve((size_t)Ll * 10 * Dd * Dd * 2);
  u16*   woB  = (u16*)  carve((size_t)Dd * Vv * 2);

  auto cvt = [&](const float* src, u16* dst, size_t n) {
    int blocks = (int)((n + 2047) / 2048);
    cvt_f32_bf16<<<blocks, 256, 0, stream>>>(src, dst, (int)n);
  };

  // ---- weight / encoder conversion (bf16) ----
  cvt(enc,  encB, (size_t)Bb * SEe * Dd);
  cvt(Wout, woB,  (size_t)Dd * Vv);
  const float* wsrc[10] = {Wq_bot, Wk_bot, Wv_bot, Wo_bot,
                           Wq_mid, Wk_mid, Wv_mid, Wo_mid, W1, W2};
  for (int i = 0; i < Ll; i++)
    for (int m = 0; m < 10; m++)
      cvt(wsrc[m] + (size_t)i * Dd * Dd, wB + ((size_t)i * 10 + m) * Dd * Dd,
          (size_t)Dd * Dd);

  // ---- embedding ----
  embed_kernel<<<Bb * Ss, 128, 0, stream>>>(seq, emb, pos, xF, xB);

  const int Mx = Bb * Ss;      // 8192
  const int Me = Bb * SEe;     // 32768

  auto gemmQKV = [&](const u16* Xp, const u16* Wp, const float* bp, u16* outb, int M) {
    dim3 g(M / 128, Dd / 128);
    gemm_kernel<0><<<g, 256, 0, stream>>>(Xp, Wp, bp, nullptr, nullptr, nullptr,
                                          nullptr, nullptr, nullptr, outb, M, Dd, Dd);
  };
  auto gemmBN = [&](const u16* Xp, const u16* Wp, const float* bp,
                    const float* gm, const float* bt, const float* mn, const float* vr) {
    dim3 g(Mx / 128, Dd / 128);
    gemm_kernel<1><<<g, 256, 0, stream>>>(Xp, Wp, bp, xF, gm, bt, mn, vr,
                                          xF, xB, Mx, Dd, Dd);
  };

  for (int i = 0; i < Ll; i++) {
    const u16* wq  = wB + ((size_t)i * 10 + 0) * Dd * Dd;
    const u16* wk  = wB + ((size_t)i * 10 + 1) * Dd * Dd;
    const u16* wv  = wB + ((size_t)i * 10 + 2) * Dd * Dd;
    const u16* wo  = wB + ((size_t)i * 10 + 3) * Dd * Dd;
    const u16* wqm = wB + ((size_t)i * 10 + 4) * Dd * Dd;
    const u16* wkm = wB + ((size_t)i * 10 + 5) * Dd * Dd;
    const u16* wvm = wB + ((size_t)i * 10 + 6) * Dd * Dd;
    const u16* wom = wB + ((size_t)i * 10 + 7) * Dd * Dd;
    const u16* w1l = wB + ((size_t)i * 10 + 8) * Dd * Dd;
    const u16* w2l = wB + ((size_t)i * 10 + 9) * Dd * Dd;

    // ---- causal self-attention ----
    gemmQKV(xB, wq, bq_bot + (size_t)i * Dd, qB, Mx);
    gemmQKV(xB, wk, bk_bot + (size_t)i * Dd, kB, Mx);
    gemmQKV(xB, wv, bv_bot + (size_t)i * Dd, vB, Mx);
    attn_kernel<true><<<dim3(Bb * Hh, Ss / 64), 128, 0, stream>>>(qB, kB, vB, aB, Ss, Ss);
    gemmBN(aB, wo, bo_bot + (size_t)i * Dd,
           gamma_bot + (size_t)i * Dd, beta_bot + (size_t)i * Dd,
           mean_bot + (size_t)i * Dd,  var_bot + (size_t)i * Dd);

    // ---- cross-attention vs encoder output ----
    gemmQKV(xB,   wqm, bq_mid + (size_t)i * Dd, qB, Mx);
    gemmQKV(encB, wkm, bk_mid + (size_t)i * Dd, kB, Me);
    gemmQKV(encB, wvm, bv_mid + (size_t)i * Dd, vB, Me);
    attn_kernel<false><<<dim3(Bb * Hh, Ss / 64), 128, 0, stream>>>(qB, kB, vB, aB, Ss, SEe);
    gemmBN(aB, wom, bo_mid + (size_t)i * Dd,
           gamma_mid + (size_t)i * Dd, beta_mid + (size_t)i * Dd,
           mean_mid + (size_t)i * Dd,  var_mid + (size_t)i * Dd);

    // ---- FFN ----
    {
      dim3 g(Mx / 128, 512 / 128);
      gemm_kernel<2><<<g, 256, 0, stream>>>(xB, w1l, b1 + (size_t)i * 512,
                                            nullptr, nullptr, nullptr, nullptr, nullptr,
                                            nullptr, mB, Mx, 512, Dd);
    }
    gemmBN(mB, w2l, b2 + (size_t)i * Dd,
           gamma_ffn + (size_t)i * Dd, beta_ffn + (size_t)i * Dd,
           mean_ffn + (size_t)i * Dd,  var_ffn + (size_t)i * Dd);
  }

  // ---- final logits: [8192,512] x [512,8000] -> f32 d_out ----
  {
    dim3 g(Mx / 128, (Vv + 127) / 128);
    gemm_kernel<3><<<g, 256, 0, stream>>>(xB, woB, bout,
                                          nullptr, nullptr, nullptr, nullptr, nullptr,
                                          (float*)d_out, nullptr, Mx, Vv, Dd);
  }
}
